// Generator_21792664059988
// MI455X (gfx1250) — matrix operand order repaired and checked by reading, not compiled, not run on hardware
//
#include <hip/hip_runtime.h>

// ---------------------------------------------------------------------------
// Autoregressive LSTM soft-embedding decoder for MI455X (gfx1250).
// bf16 WMMA (v_wmma_f32_16x16x32_bf16) with fp32 accumulation + fp32 state.
// Weights converted once per launch into B-operand tile layout (L2-resident).
// ---------------------------------------------------------------------------

#define B_    32
#define T_    64
#define DEMB  512
#define DCONT 128
#define DSTY  16
#define DIN   656          // DEMB + DCONT + DSTY
#define DINP  672          // DIN padded to multiple of 32
#define VOCAB 32000
#define NGATE 2048         // 4 * DEMB
#define KC_K  (DINP / 32)  // 21 K-chunks for x@Wk
#define KC_R  (DEMB / 32)  // 16 K-chunks for h@Wr
#define KC_S  (DEMB / 32)  // 16 K-chunks for h@Ws
#define KC_E  (VOCAB / 32) // 1000 K-chunks for p@E
#define NSLICE 10          // split-K slices for p@E (deterministic reduce)
#define KPS   (KC_E / NSLICE) // 100 chunks per slice

typedef __bf16 bf16x16 __attribute__((ext_vector_type(16)));
typedef __bf16 bf16x8  __attribute__((ext_vector_type(8)));
typedef float  f32x8   __attribute__((ext_vector_type(8)));

static __device__ __forceinline__ f32x8 wmma_bf16(bf16x16 a, bf16x16 b, f32x8 c) {
  // D = A(16x32) * B(32x16) + C(16x16 f32)
  return __builtin_amdgcn_wmma_f32_16x16x32_bf16(false, a, false, b, (short)0, c,
                                                 false, false);
}

// A operand (16-bit 16x32): lane<16 -> row=lane, K 0..7 then 16..23;
// lane>=16 -> row=lane-16, K 8..15 then 24..31. From row-major [M][Kpad] this
// is two contiguous, 16B-aligned 8xbf16 loads at p and p+16.
static __device__ __forceinline__ bf16x16 load_a16x32(const __bf16* p) {
  bf16x8 lo = *(const bf16x8*)p;
  bf16x8 hi = *(const bf16x8*)(p + 16);
  return __builtin_shufflevector(lo, hi, 0, 1, 2, 3, 4, 5, 6, 7,
                                         8, 9, 10, 11, 12, 13, 14, 15);
}

// ---------------------------------------------------------------------------
// One-time weight conversion: fp32 row-major [K][N] -> bf16 B-operand tiles.
// Block (nt, kc) is 512 elems: lane ln's 16 bf16 at offset lane*16 hold
// W[kc*32 + (lane>>4)*8 + (e&7) + (e>>3)*16][nt*16 + (lane&15)].  (k >= K zero-padded)
// ---------------------------------------------------------------------------
__global__ void convert_bt_kernel(const float* __restrict__ src,
                                  __bf16* __restrict__ dst,
                                  int K, int Kpad, int N) {
  long long id = (long long)blockIdx.x * blockDim.x + threadIdx.x;
  long long total = (long long)Kpad * N;
  if (id >= total) return;
  int e    = (int)(id & 15);
  int lane = (int)((id >> 4) & 31);
  long long blk = id >> 9;
  int KC = Kpad >> 5;
  int kc = (int)(blk % KC);
  int nt = (int)(blk / KC);
  int k = kc * 32 + (lane >> 4) * 8 + (e & 7) + (e >> 3) * 16;
  int n = nt * 16 + (lane & 15);
  float v = (k < K) ? src[(long long)k * N + n] : 0.0f;
  dst[id] = (__bf16)v;
}

// ---------------------------------------------------------------------------
// Init: x = [E[SOS], content, style, pad], h=c=0, out[:,0,:] = E[SOS].
// ---------------------------------------------------------------------------
__global__ void init_state_kernel(const float* __restrict__ E,
                                  const float* __restrict__ content,
                                  const float* __restrict__ style,
                                  __bf16* __restrict__ xb,
                                  __bf16* __restrict__ hbA,
                                  __bf16* __restrict__ hbB,
                                  float* __restrict__ c,
                                  float* __restrict__ out) {
  int id = blockIdx.x * blockDim.x + threadIdx.x;
  const int NXB = B_ * DINP;  // 21504
  if (id < NXB) {
    int b = id / DINP, col = id % DINP;
    float v;
    if (col < DEMB)              v = E[col];  // E[SOS=0][col]
    else if (col < DEMB + DCONT) v = content[b * DCONT + (col - DEMB)];
    else if (col < DIN)          v = style[b * DSTY + (col - DEMB - DCONT)];
    else                         v = 0.0f;
    xb[id] = (__bf16)v;
  } else {
    int r = id - NXB;
    if (r < B_ * DEMB) {
      int b = r >> 9, d = r & 511;
      hbA[r] = (__bf16)0.0f;
      hbB[r] = (__bf16)0.0f;
      c[r] = 0.0f;
      out[(long long)b * (T_ * DEMB) + d] = E[d];  // t = 0
    }
  }
}

// ---------------------------------------------------------------------------
// Fused LSTM gates: z = x@Wk + h@Wr + b, then pointwise i/f/g/o update.
// 32 blocks x 128 threads.  Wave g computes gate g's 32x16 tile for unit
// tile j=blockIdx.x (z columns 512*g + 16*j .. +16).  Gates meet in LDS.
// ---------------------------------------------------------------------------
__global__ void lstm_gates_kernel(const __bf16* __restrict__ xb,
                                  const __bf16* __restrict__ hin,
                                  const __bf16* __restrict__ wkb,
                                  const __bf16* __restrict__ wrb,
                                  const float* __restrict__ bias,
                                  float* __restrict__ c,
                                  __bf16* __restrict__ hout) {
  __shared__ float zs[4][B_][16];
  const int lane  = threadIdx.x & 31;
  const int g     = threadIdx.x >> 5;  // gate 0..3 (i,f,g,o)
  const int j     = blockIdx.x;        // unit tile 0..31
  const int nt    = g * 32 + j;        // n-tile within [0,2048)
  const int khalf = lane >> 4;
  const int ln    = lane & 15;
  f32x8 acc0 = {0, 0, 0, 0, 0, 0, 0, 0};
  f32x8 acc1 = {0, 0, 0, 0, 0, 0, 0, 0};
  {  // x @ Wk  (Kpad = 672)
    const __bf16* bblk = wkb + (long long)nt * KC_K * 512;
    for (int kc = 0; kc < KC_K; ++kc) {
      bf16x16 bv = *(const bf16x16*)(bblk + kc * 512 + lane * 16);
      const __bf16* ap = xb + ln * DINP + kc * 32 + khalf * 8;
      acc0 = wmma_bf16(load_a16x32(ap), bv, acc0);
      acc1 = wmma_bf16(load_a16x32(ap + 16 * DINP), bv, acc1);
    }
  }
  {  // h @ Wr  (K = 512)
    const __bf16* bblk = wrb + (long long)nt * KC_R * 512;
    for (int kc = 0; kc < KC_R; ++kc) {
      bf16x16 bv = *(const bf16x16*)(bblk + kc * 512 + lane * 16);
      const __bf16* ap = hin + ln * DEMB + kc * 32 + khalf * 8;
      acc0 = wmma_bf16(load_a16x32(ap), bv, acc0);
      acc1 = wmma_bf16(load_a16x32(ap + 16 * DEMB), bv, acc1);
    }
  }
  // C/D layout: lane<16 -> (row=v, col=lane); lane>=16 -> (row=v+8, col=lane-16)
#pragma unroll
  for (int v = 0; v < 8; ++v) {
    zs[g][v + 8 * khalf][ln]      = acc0[v];
    zs[g][16 + v + 8 * khalf][ln] = acc1[v];
  }
  __syncthreads();
#pragma unroll
  for (int rr = 0; rr < 4; ++rr) {
    int e = threadIdx.x + 128 * rr;  // 0..511 = (batch, unit-in-tile)
    int b = e >> 4, u = e & 15;
    int unit = j * 16 + u;
    float zi = zs[0][b][u] + bias[unit];
    float zf = zs[1][b][u] + bias[512 + unit];
    float zg = zs[2][b][u] + bias[1024 + unit];
    float zo = zs[3][b][u] + bias[1536 + unit];
    float ig = 1.0f / (1.0f + __expf(-zi));
    float fg = 1.0f / (1.0f + __expf(-zf));
    float gg = tanhf(zg);
    float og = 1.0f / (1.0f + __expf(-zo));
    int idx = b * DEMB + unit;
    float cc = fg * c[idx] + ig * gg;
    c[idx] = cc;
    hout[idx] = (__bf16)(og * tanhf(cc));
  }
}

// ---------------------------------------------------------------------------
// logits[32,32000] = h @ Ws + bs.   500 blocks x 128 thr; one n-tile per wave.
// ---------------------------------------------------------------------------
__global__ void logits_kernel(const __bf16* __restrict__ hb,
                              const __bf16* __restrict__ wsb,
                              const float* __restrict__ bs,
                              float* __restrict__ logits) {
  const int lane  = threadIdx.x & 31;
  const int nt    = blockIdx.x * 4 + (threadIdx.x >> 5);  // 0..1999
  const int khalf = lane >> 4;
  const int ln    = lane & 15;
  f32x8 acc0 = {0, 0, 0, 0, 0, 0, 0, 0};
  f32x8 acc1 = {0, 0, 0, 0, 0, 0, 0, 0};
  const __bf16* bblk = wsb + (long long)nt * KC_S * 512;
  for (int kc = 0; kc < KC_S; ++kc) {
    bf16x16 bv = *(const bf16x16*)(bblk + kc * 512 + lane * 16);
    const __bf16* ap = hb + ln * DEMB + kc * 32 + khalf * 8;
    acc0 = wmma_bf16(load_a16x32(ap), bv, acc0);
    acc1 = wmma_bf16(load_a16x32(ap + 16 * DEMB), bv, acc1);
  }
  int n = nt * 16 + ln;
  float bb = bs[n];
#pragma unroll
  for (int v = 0; v < 8; ++v) {
    logits[(long long)(v + 8 * khalf) * VOCAB + n]      = acc0[v] + bb;
    logits[(long long)(16 + v + 8 * khalf) * VOCAB + n] = acc1[v] + bb;
  }
}

// ---------------------------------------------------------------------------
// Row softmax over 32000; exp cached in-place in logits; p written bf16.
// ---------------------------------------------------------------------------
__global__ void softmax_kernel(float* __restrict__ logits,
                               __bf16* __restrict__ pb) {
  __shared__ float red[256];
  const int row = blockIdx.x, tid = threadIdx.x;
  float* lrow = logits + (long long)row * VOCAB;
  float m = -3.4e38f;
  for (int i = tid; i < VOCAB; i += 256) m = fmaxf(m, lrow[i]);
  red[tid] = m;
  __syncthreads();
  for (int s = 128; s > 0; s >>= 1) {
    if (tid < s) red[tid] = fmaxf(red[tid], red[tid + s]);
    __syncthreads();
  }
  m = red[0];
  __syncthreads();
  float sum = 0.0f;
  for (int i = tid; i < VOCAB; i += 256) {
    float e = __expf(lrow[i] - m);
    lrow[i] = e;
    sum += e;
  }
  red[tid] = sum;
  __syncthreads();
  for (int s = 128; s > 0; s >>= 1) {
    if (tid < s) red[tid] += red[tid + s];
    __syncthreads();
  }
  float inv = 1.0f / red[0];
  __bf16* prow = pb + (long long)row * VOCAB;
  for (int i = tid; i < VOCAB; i += 256) prow[i] = (__bf16)(lrow[i] * inv);
}

// ---------------------------------------------------------------------------
// emb = p @ E, split-K into NSLICE deterministic partials. 80 blocks x 128.
// wave w: nt = w&31 (n-tile of 512), slice = w>>5 (100 K-chunks each).
// ---------------------------------------------------------------------------
__global__ void emb_gemm_kernel(const __bf16* __restrict__ pb,
                                const __bf16* __restrict__ eb,
                                float* __restrict__ partials) {
  const int lane  = threadIdx.x & 31;
  const int w     = blockIdx.x * 4 + (threadIdx.x >> 5);  // 0..319
  const int nt    = w & 31;
  const int slice = w >> 5;  // 0..9
  const int khalf = lane >> 4;
  const int ln    = lane & 15;
  f32x8 acc0 = {0, 0, 0, 0, 0, 0, 0, 0};
  f32x8 acc1 = {0, 0, 0, 0, 0, 0, 0, 0};
  const __bf16* bblk = eb + (long long)nt * KC_E * 512;
  const int kc0 = slice * KPS;
  for (int kc = kc0; kc < kc0 + KPS; ++kc) {
    bf16x16 bv = *(const bf16x16*)(bblk + (long long)kc * 512 + lane * 16);
    const __bf16* ap = pb + (long long)ln * VOCAB + kc * 32 + khalf * 8;
    acc0 = wmma_bf16(load_a16x32(ap), bv, acc0);
    acc1 = wmma_bf16(load_a16x32(ap + (long long)16 * VOCAB), bv, acc1);
  }
#pragma unroll
  for (int v = 0; v < 8; ++v) {
    partials[(slice * B_ + (v + 8 * khalf)) * DEMB + nt * 16 + ln]      = acc0[v];
    partials[(slice * B_ + (16 + v + 8 * khalf)) * DEMB + nt * 16 + ln] = acc1[v];
  }
}

// ---------------------------------------------------------------------------
// Reduce split-K partials; write out[:,t,:] (fp32) and x emb part (bf16).
// ---------------------------------------------------------------------------
__global__ void emb_reduce_kernel(const float* __restrict__ partials,
                                  float* __restrict__ out_t,
                                  __bf16* __restrict__ xb) {
  int id = blockIdx.x * blockDim.x + threadIdx.x;  // 0..16383
  int b = id >> 9, d = id & 511;
  float s = 0.0f;
#pragma unroll
  for (int sl = 0; sl < NSLICE; ++sl) s += partials[(sl * B_ + b) * DEMB + d];
  out_t[(long long)b * (T_ * DEMB) + d] = s;
  xb[b * DINP + d] = (__bf16)s;
}

// ---------------------------------------------------------------------------
extern "C" void kernel_launch(void* const* d_in, const int* in_sizes, int n_in,
                              void* d_out, int out_size, void* d_ws, size_t ws_size,
                              hipStream_t stream) {
  const float* content = (const float*)d_in[0];
  const float* style   = (const float*)d_in[1];
  const float* E       = (const float*)d_in[2];
  const float* Wk      = (const float*)d_in[3];
  const float* Wr      = (const float*)d_in[4];
  const float* bias    = (const float*)d_in[5];
  const float* Ws      = (const float*)d_in[6];
  const float* bs      = (const float*)d_in[7];
  float* out = (float*)d_out;

  char* ws = (char*)d_ws;
  size_t off = 0;
  auto carve = [&](size_t bytes) -> char* {
    off = (off + 255) & ~(size_t)255;
    char* p = ws + off;
    off += bytes;
    return p;
  };
  __bf16* wkb = (__bf16*)carve((size_t)DINP * NGATE * 2);   // 2.75 MB
  __bf16* wrb = (__bf16*)carve((size_t)DEMB * NGATE * 2);   // 2.10 MB
  __bf16* wsb = (__bf16*)carve((size_t)DEMB * VOCAB * 2);   // 32.8 MB
  __bf16* eb  = (__bf16*)carve((size_t)VOCAB * DEMB * 2);   // 32.8 MB
  __bf16* xb  = (__bf16*)carve((size_t)B_ * DINP * 2);
  __bf16* hbA = (__bf16*)carve((size_t)B_ * DEMB * 2);
  __bf16* hbB = (__bf16*)carve((size_t)B_ * DEMB * 2);
  float*  cst = (float*)carve((size_t)B_ * DEMB * 4);
  float*  logits   = (float*)carve((size_t)B_ * VOCAB * 4); // 4.1 MB
  __bf16* pb       = (__bf16*)carve((size_t)B_ * VOCAB * 2);
  float*  partials = (float*)carve((size_t)NSLICE * B_ * DEMB * 4);
  if (off > ws_size) return;  // workspace too small: cannot proceed safely

  // --- one-time (per launch) weight conversion into B-operand tile layout ---
  {
    long long tot;
    tot = (long long)DINP * NGATE;   // 1,376,256
    convert_bt_kernel<<<(int)(tot / 256), 256, 0, stream>>>(Wk, wkb, DIN, DINP, NGATE);
    tot = (long long)DEMB * NGATE;   // 1,048,576
    convert_bt_kernel<<<(int)(tot / 256), 256, 0, stream>>>(Wr, wrb, DEMB, DEMB, NGATE);
    tot = (long long)DEMB * VOCAB;   // 16,384,000
    convert_bt_kernel<<<(int)(tot / 256), 256, 0, stream>>>(Ws, wsb, DEMB, DEMB, VOCAB);
    tot = (long long)VOCAB * DEMB;   // 16,384,000
    convert_bt_kernel<<<(int)(tot / 256), 256, 0, stream>>>(E, eb, VOCAB, VOCAB, DEMB);
  }
  init_state_kernel<<<148, 256, 0, stream>>>(E, content, style, xb, hbA, hbB, cst, out);

  // --- 63 autoregressive steps (all weights hot in 192 MB L2 after step 1) ---
  for (int t = 1; t < T_; ++t) {
    const __bf16* hin = (t & 1) ? hbA : hbB;  // double-buffered h (RAW across WGs)
    __bf16* hout      = (t & 1) ? hbB : hbA;
    lstm_gates_kernel<<<32, 128, 0, stream>>>(xb, hin, wkb, wrb, bias, cst, hout);
    logits_kernel<<<500, 128, 0, stream>>>(hout, wsb, bs, logits);
    softmax_kernel<<<32, 256, 0, stream>>>(logits, pb);
    emb_gemm_kernel<<<80, 128, 0, stream>>>(pb, eb, partials);
    emb_reduce_kernel<<<64, 256, 0, stream>>>(partials, out + (size_t)t * DEMB, xb);
  }
  (void)in_sizes; (void)n_in; (void)out_size;
}
